// GAT_HO_19344532701815
// MI455X (gfx1250) — matrix operand order repaired and checked by reading, not compiled, not run on hardware
//
#include <hip/hip_runtime.h>
#include <hip/hip_bf16.h>

#define N_NODES 50000
#define N_EDGES 600000
#define DIM 128
#define N_HEADS 3
#define NEG_SLOPE 0.2f

typedef __attribute__((ext_vector_type(16))) __bf16 v16bf;
typedef __attribute__((ext_vector_type(8)))  float  v8f;

#define LDX 136   // 128 + 8 bf16 pad (stagger LDS banks)
#define LDW 136

// ---------------------------------------------------------------- utilities
__global__ void fill_u32_kernel(unsigned int* p, unsigned int v, int n) {
    int i = blockIdx.x * blockDim.x + threadIdx.x;
    if (i < n) p[i] = v;
}

__global__ void ew_kernel(const int* __restrict__ edge_ids,
                          const float* __restrict__ ddi,
                          const float* __restrict__ emb,
                          float* __restrict__ ew) {
    int i = blockIdx.x * blockDim.x + threadIdx.x;
    if (i < N_EDGES) ew[i] = emb[edge_ids[i]] - ddi[i];
}

// order-preserving float<->uint encoding for atomicMax over floats
__device__ __forceinline__ unsigned enc_f(float f) {
    unsigned u = __float_as_uint(f);
    return (u & 0x80000000u) ? ~u : (u | 0x80000000u);
}
__device__ __forceinline__ float dec_f(unsigned e) {
    unsigned u = (e & 0x80000000u) ? (e ^ 0x80000000u) : ~e;
    return __uint_as_float(u);
}

// ---------------------------------------------------------------- GEMM (WMMA)
// H = cvt_bf16(X[Nx128]) @ cvt_bf16(W[128x128]) (+ bias), f32 accumulate.
// Block: 128 threads = 4 waves; each block does a 64-row M tile; each wave a
// 16(M) x 128(N) strip via 8 accumulators, K-loop of 4 x 32.
__global__ __launch_bounds__(128) void gemm_wmma_kernel(
        const float* __restrict__ X, const float* __restrict__ W,
        const float* __restrict__ bias, int has_bias, float* __restrict__ H) {
    __shared__ __bf16 Xs[64 * LDX];     // X tile, row-major [m][k]
    __shared__ __bf16 Wt[128 * LDW];    // W transposed    [n][k]
    const int t  = threadIdx.x;
    const int m0 = blockIdx.x * 64;

    // stage X tile (64x128 f32 -> bf16); 2048 float4 slots / 128 threads
    for (int i = 0; i < 16; ++i) {
        int idx = t + i * 128;
        int row = idx >> 5;
        int c4  = idx & 31;
        float4 f = make_float4(0.f, 0.f, 0.f, 0.f);
        int gr = m0 + row;
        if (gr < N_NODES) f = ((const float4*)(X + (size_t)gr * DIM))[c4];
        __bf16* d = &Xs[row * LDX + c4 * 4];
        d[0] = (__bf16)f.x; d[1] = (__bf16)f.y; d[2] = (__bf16)f.z; d[3] = (__bf16)f.w;
    }
    // stage W transposed: thread t reads column t (coalesced over t per row)
    for (int i = 0; i < 128; ++i) {
        Wt[t * LDW + i] = (__bf16)W[(size_t)i * DIM + t];
    }
    __syncthreads();

    const int wv   = t >> 5;
    const int lane = t & 31;
    const int hi   = lane >> 4;   // half-wave select (ISA A/B lane layout)
    const int lm   = lane & 15;

    v8f acc[8] = {};
    const __bf16* xrow = &Xs[(wv * 16 + lm) * LDX];

    for (int ks = 0; ks < 4; ++ks) {
        const int kb = ks * 32;
        // A fragment: 16x32 bf16. lanes 0-15: K = kb+{0..15};
        //             lanes 16-31: K = kb+{16..31}
        v16bf a;
#pragma unroll
        for (int j = 0; j < 8; ++j) {
            a[j]     = xrow[kb + hi * 16 + j];
            a[8 + j] = xrow[kb + hi * 16 + 8 + j];
        }
#pragma unroll
        for (int nt = 0; nt < 8; ++nt) {
            // B fragment: 32x16 bf16. lane holds column n = nt*16+lm,
            // K = kb + hi*16 + e  (contiguous 16 bf16 from transposed W)
            const __bf16* wrow = &Wt[(nt * 16 + lm) * LDW + kb + hi * 16];
            v16bf b;
#pragma unroll
            for (int e = 0; e < 16; ++e) b[e] = wrow[e];
            acc[nt] = __builtin_amdgcn_wmma_f32_16x16x32_bf16(
                false, a, false, b, (short)0, acc[nt], false, false);
        }
    }

    // epilogue: D layout -> VGPR r holds rows r (lanes 0-15) / r+8 (16-31)
#pragma unroll
    for (int nt = 0; nt < 8; ++nt) {
        int col  = nt * 16 + lm;
        float bv = has_bias ? bias[col] : 0.f;
#pragma unroll
        for (int r = 0; r < 8; ++r) {
            int m = m0 + wv * 16 + hi * 8 + r;
            if (m < N_NODES) H[(size_t)m * DIM + col] = acc[nt][r] + bv;
        }
    }
}

// ---------------------------------------------------------------- scores
// one wave per node: ssrc[n] = h[n].a_src, sdst[n] = h[n].a_dst
__global__ void scores_kernel(const float* __restrict__ H,
                              const float* __restrict__ as,
                              const float* __restrict__ ad,
                              float* __restrict__ ssrc, float* __restrict__ sdst) {
    int node = (int)((blockIdx.x * blockDim.x + threadIdx.x) >> 5);
    int lane = threadIdx.x & 31;
    if (node >= N_NODES) return;
    float4 h4 = ((const float4*)(H + (size_t)node * DIM))[lane];
    float4 a4 = ((const float4*)as)[lane];
    float4 d4 = ((const float4*)ad)[lane];
    float ps = h4.x * a4.x + h4.y * a4.y + h4.z * a4.z + h4.w * a4.w;
    float pd = h4.x * d4.x + h4.y * d4.y + h4.z * d4.z + h4.w * d4.w;
    for (int off = 16; off > 0; off >>= 1) {
        ps += __shfl_down(ps, off, 32);
        pd += __shfl_down(pd, off, 32);
    }
    if (lane == 0) { ssrc[node] = ps; sdst[node] = pd; }
}

// ---------------------------------------------------------------- edge passes
__global__ void edge_pass1_kernel(const int* __restrict__ ei,
                                  const float* __restrict__ ssrc,
                                  const float* __restrict__ sdst,
                                  float* __restrict__ ebuf, unsigned* __restrict__ emax) {
    int i = blockIdx.x * blockDim.x + threadIdx.x;
    if (i >= N_EDGES) return;
    int s = ei[i], d = ei[N_EDGES + i];
    float e = ssrc[s] + sdst[d];
    e = (e >= 0.f) ? e : NEG_SLOPE * e;
    ebuf[i] = e;
    atomicMax(&emax[d], enc_f(e));
}

__global__ void edge_pass2_kernel(const int* __restrict__ ei,
                                  const unsigned* __restrict__ emax,
                                  float* __restrict__ ebuf, float* __restrict__ denom) {
    int i = blockIdx.x * blockDim.x + threadIdx.x;
    if (i >= N_EDGES) return;
    int d = ei[N_EDGES + i];
    float x = __expf(ebuf[i] - dec_f(emax[d]));
    ebuf[i] = x;
    atomicAdd(&denom[d], x);
}

// one wave per edge; lane covers 4 of 128 channels
__global__ void edge_pass3_kernel(const int* __restrict__ ei,
                                  const float* __restrict__ ebuf,
                                  const float* __restrict__ denom,
                                  const float* __restrict__ ew,
                                  const float* __restrict__ H, float* __restrict__ out) {
    int e    = (int)((blockIdx.x * blockDim.x + threadIdx.x) >> 5);
    int lane = threadIdx.x & 31;
    if (e >= N_EDGES) return;
    int s = ei[e], d = ei[N_EDGES + e];
    float coef = (ebuf[e] / fmaxf(denom[d], 1e-16f)) * ew[e];
    const float4* hp = (const float4*)(H + (size_t)s * DIM);
    __builtin_prefetch(hp + lane, 0, 0);         // global_prefetch_b8
    float4 h4 = hp[lane];
    float* op = out + (size_t)d * DIM + lane * 4;
    atomicAdd(op + 0, coef * h4.x);
    atomicAdd(op + 1, coef * h4.y);
    atomicAdd(op + 2, coef * h4.z);
    atomicAdd(op + 3, coef * h4.w);
}

__global__ void finalize_kernel(float* __restrict__ out, const float* __restrict__ bh) {
    int i = blockIdx.x * blockDim.x + threadIdx.x;
    if (i >= N_NODES * DIM) return;
    int c = i & (DIM - 1);
    float b = bh[c] + bh[DIM + c] + bh[2 * DIM + c];
    out[i] = (out[i] + b) * (1.f / 3.f);
}

// ---------------------------------------------------------------- launch
extern "C" void kernel_launch(void* const* d_in, const int* in_sizes, int n_in,
                              void* d_out, int out_size, void* d_ws, size_t ws_size,
                              hipStream_t stream) {
    const float* x          = (const float*)d_in[0];
    const int*   edge_index = (const int*)d_in[1];
    const int*   edge_ids   = (const int*)d_in[2];
    const float* ddi        = (const float*)d_in[3];
    const float* W_lin      = (const float*)d_in[4];
    const float* b_lin      = (const float*)d_in[5];
    const float* edge_emb   = (const float*)d_in[6];
    const float* W_heads    = (const float*)d_in[7];
    const float* att_src    = (const float*)d_in[8];
    const float* att_dst    = (const float*)d_in[9];
    const float* bias_heads = (const float*)d_in[10];

    char* ws = (char*)d_ws;
    float*    Xp    = (float*)ws;    ws += (size_t)N_NODES * DIM * 4;
    float*    H     = (float*)ws;    ws += (size_t)N_NODES * DIM * 4;
    float*    ssrc  = (float*)ws;    ws += (size_t)N_NODES * 4;
    float*    sdst  = (float*)ws;    ws += (size_t)N_NODES * 4;
    unsigned* emax  = (unsigned*)ws; ws += (size_t)N_NODES * 4;
    float*    denom = (float*)ws;    ws += (size_t)N_NODES * 4;
    float*    ebuf  = (float*)ws;    ws += (size_t)N_EDGES * 4;
    float*    ew    = (float*)ws;

    const unsigned ENC_NEG_INF = 0x007FFFFFu;  // enc_f(-inf)

    // zero output accumulator (d_out is poisoned by harness)
    fill_u32_kernel<<<(N_NODES * DIM + 255) / 256, 256, 0, stream>>>(
        (unsigned*)d_out, 0u, N_NODES * DIM);

    // shared projection: Xp = x @ W_lin + b_lin
    gemm_wmma_kernel<<<(N_NODES + 63) / 64, 128, 0, stream>>>(x, W_lin, b_lin, 1, Xp);

    // edge weights: ew = edge_emb[edge_ids] - ddi_weight
    ew_kernel<<<(N_EDGES + 255) / 256, 256, 0, stream>>>(edge_ids, ddi, edge_emb, ew);

    for (int k = 0; k < N_HEADS; ++k) {
        gemm_wmma_kernel<<<(N_NODES + 63) / 64, 128, 0, stream>>>(
            Xp, W_heads + (size_t)k * DIM * DIM, b_lin, 0, H);
        scores_kernel<<<(N_NODES + 7) / 8, 256, 0, stream>>>(
            H, att_src + k * DIM, att_dst + k * DIM, ssrc, sdst);
        fill_u32_kernel<<<(N_NODES + 255) / 256, 256, 0, stream>>>(emax, ENC_NEG_INF, N_NODES);
        fill_u32_kernel<<<(N_NODES + 255) / 256, 256, 0, stream>>>((unsigned*)denom, 0u, N_NODES);
        edge_pass1_kernel<<<(N_EDGES + 255) / 256, 256, 0, stream>>>(
            edge_index, ssrc, sdst, ebuf, emax);
        edge_pass2_kernel<<<(N_EDGES + 255) / 256, 256, 0, stream>>>(
            edge_index, emax, ebuf, denom);
        edge_pass3_kernel<<<(N_EDGES + 7) / 8, 256, 0, stream>>>(
            edge_index, ebuf, denom, ew, H, (float*)d_out);
    }

    finalize_kernel<<<(N_NODES * DIM + 255) / 256, 256, 0, stream>>>(
        (float*)d_out, bias_heads);
}
